// LSTMDecoder_72610717106196
// MI455X (gfx1250) — compile-verified, matrix operand order checked by reading, not run on hardware
//
#include <hip/hip_runtime.h>

// ---------------- problem constants (match reference) ----------------
#define Bdim   64
#define Nimg   196
#define Dimg   2048
#define Tsteps 32
#define Vv     32000
#define Ee     512
#define Hh     1024
#define Aa     256

typedef __bf16 bf16_t;
typedef __attribute__((ext_vector_type(16))) __bf16 v16bf;
typedef __attribute__((ext_vector_type(8)))  __bf16 v8bf;
typedef __attribute__((ext_vector_type(8)))  float  v8f;

union V16U { v16bf v; v8bf h[2]; };

// ---------------- fp32 -> bf16 conversion ----------------
__global__ void f32_to_bf16_kernel(const float* __restrict__ src,
                                   bf16_t* __restrict__ dst, int n) {
  int i = blockIdx.x * blockDim.x + threadIdx.x;
  if (i < n) dst[i] = (bf16_t)src[i];
}

// ---------------- bias_sum = b_ih + b_hh ----------------
__global__ void bias_sum_kernel(const float* __restrict__ a,
                                const float* __restrict__ b,
                                float* __restrict__ o, int n) {
  int i = blockIdx.x * blockDim.x + threadIdx.x;
  if (i < n) o[i] = a[i] + b[i];
}

// ---------------- g = mean_n(img_seq) ----------------
__global__ void mean_kernel(const float* __restrict__ img, float* __restrict__ g) {
  int idx = blockIdx.x * blockDim.x + threadIdx.x;      // over B*Dimg
  int b = idx >> 11, d = idx & (Dimg - 1);
  const float* p = img + (size_t)b * Nimg * Dimg + d;
  float s = 0.f;
  for (int n = 0; n < Nimg; ++n) s += p[(size_t)n * Dimg];
  g[idx] = s * (1.0f / (float)Nimg);
}

// ---------------- h0 / c0 = tanh(g @ W^T + b) ----------------
__global__ void init_hc_kernel(const float* __restrict__ g,
                               const float* __restrict__ W_h0, const float* __restrict__ b_h0,
                               const float* __restrict__ W_c0, const float* __restrict__ b_c0,
                               float* __restrict__ h, float* __restrict__ c,
                               bf16_t* __restrict__ h_bf) {
  int idx = blockIdx.x * blockDim.x + threadIdx.x;      // over B*Hh
  int which = blockIdx.y;                               // 0 = h, 1 = c
  int b = idx >> 10, j = idx & (Hh - 1);
  const float* W  = which ? W_c0 : W_h0;
  const float* bb = which ? b_c0 : b_h0;
  const float* gb = g + (size_t)b * Dimg;
  const float* wr = W + (size_t)j * Dimg;
  float s = bb[j];
  for (int k = 0; k < Dimg; ++k) s += gb[k] * wr[k];
  s = tanhf(s);
  if (which) { c[idx] = s; }
  else       { h[idx] = s; h_bf[idx] = (bf16_t)s; }
}

// ---------------- generic WMMA GEMM, 16x64 wave tile ----------------
// C[M,N](f32) = A[M,K](bf16) @ B[N,K](bf16)^T  (+bias[n] | +=C)
// block = 4 waves covering 4 M-tiles; each wave owns a 16x64 output strip
// (4 accumulators) so one A fragment feeds 4 WMMAs per K-chunk.
__global__ void __launch_bounds__(128)
wmma_gemm_bf16_kernel(const bf16_t* __restrict__ A, int lda,
                      const bf16_t* __restrict__ B, int ldb,
                      float* __restrict__ outF, int ldo,
                      bf16_t* __restrict__ outBF, int ldobf,
                      const float* __restrict__ bias,
                      int M, int N, int K, int accum) {
  const int lane = threadIdx.x & 31;
  const int wave = threadIdx.x >> 5;
  const int mt = blockIdx.y * 4 + wave;
  const int m0 = mt * 16;
  const int n0 = blockIdx.x * 64;      // 4 n-subtiles of 16
  if (m0 >= M) return;
  const int hf = lane >> 4;        // half-wave selects K octet (A/B), M octet (C/D)
  const int mr = lane & 15;        // A: row in tile, B: col in tile, C/D: col in tile

  const bf16_t* arow  = A + (size_t)(m0 + mr) * lda;
  const bf16_t* brow0 = B + (size_t)(n0 + mr) * ldb;   // subtile j: + j*16*ldb

  v8f acc[4];
  if (accum) {
    for (int j = 0; j < 4; ++j)
      for (int r = 0; r < 8; ++r)
        acc[j][r] = outF[(size_t)(m0 + r + 8 * hf) * ldo + n0 + 16 * j + mr];
  } else if (bias) {
    for (int j = 0; j < 4; ++j) {
      const float bv = bias[n0 + 16 * j + mr];
      for (int r = 0; r < 8; ++r) acc[j][r] = bv;
    }
  } else {
    for (int j = 0; j < 4; ++j)
      for (int r = 0; r < 8; ++r) acc[j][r] = 0.f;
  }

  for (int k0 = 0; k0 < K; k0 += 32) {
    // ISA 16-bit A 16x32 layout: elems 0..7 <- K = k0+8*hf .. +7,
    //                            elems 8..15 <- K = k0+16+8*hf .. +7
    V16U ua;
    ua.h[0] = *(const v8bf*)(arow + k0 + 8 * hf);
    ua.h[1] = *(const v8bf*)(arow + k0 + 16 + 8 * hf);
    if (k0 + 32 < K) {   // global_prefetch_b8 of next K chunk
      __builtin_prefetch((const void*)(arow + k0 + 32), 0, 1);
      __builtin_prefetch((const void*)(brow0 + k0 + 32), 0, 1);
      __builtin_prefetch((const void*)(brow0 + (size_t)32 * ldb + k0 + 32), 0, 1);
    }
#pragma unroll
    for (int j = 0; j < 4; ++j) {
      const bf16_t* br = brow0 + (size_t)j * 16 * ldb;
      V16U ub;
      ub.h[0] = *(const v8bf*)(br + k0 + 8 * hf);
      ub.h[1] = *(const v8bf*)(br + k0 + 16 + 8 * hf);
      acc[j] = __builtin_amdgcn_wmma_f32_16x16x32_bf16(false, ua.v, false, ub.v,
                                                       (short)0, acc[j], false, false);
    }
  }

  for (int j = 0; j < 4; ++j) {
    for (int r = 0; r < 8; ++r) {
      const int m = m0 + r + 8 * hf;
      const int n = n0 + 16 * j + mr;
      const float v = acc[j][r];
      if (outF)  outF[(size_t)m * ldo + n] = v;
      if (outBF) outBF[(size_t)m * ldobf + n] = (bf16_t)v;
    }
  }
}

// ---------------- attention + context + embed gather ----------------
// one block per batch element, 256 threads
__global__ void __launch_bounds__(256)
attn_ctx_kernel(const float* __restrict__ h, const float* __restrict__ W_hid,
                const float* __restrict__ w_score, const float* __restrict__ proj_img,
                const float* __restrict__ img_seq, const float* __restrict__ embed,
                const int* __restrict__ tokens, int t, bf16_t* __restrict__ xin) {
  const int b = blockIdx.x;
  const int tid = threadIdx.x;
  __shared__ float ph[Aa];
  __shared__ float sc[Nimg];
  __shared__ float red[256];

  { // proj_h = h[b] @ W_hid^T  (thread = attention unit)
    const float* hb = h + (size_t)b * Hh;
    const float* wr = W_hid + (size_t)tid * Hh;
    float s = 0.f;
    for (int k = 0; k < Hh; ++k) s += hb[k] * wr[k];
    ph[tid] = s;
  }
  __syncthreads();

  float val = -1e30f;
  if (tid < Nimg) {
    const float* pi = proj_img + ((size_t)b * Nimg + tid) * Aa;
    float s = 0.f;
    for (int a = 0; a < Aa; ++a) s += w_score[a] * tanhf(pi[a] + ph[a]);
    val = s;
  }
  red[tid] = val;
  __syncthreads();
  for (int s = 128; s > 0; s >>= 1) {
    if (tid < s) red[tid] = fmaxf(red[tid], red[tid + s]);
    __syncthreads();
  }
  const float mx = red[0];
  __syncthreads();
  const float ev = (tid < Nimg) ? __expf(val - mx) : 0.f;
  red[tid] = ev;
  __syncthreads();
  for (int s = 128; s > 0; s >>= 1) {
    if (tid < s) red[tid] += red[tid + s];
    __syncthreads();
  }
  const float inv = 1.f / red[0];
  __syncthreads();
  if (tid < Nimg) sc[tid] = ev * inv;
  __syncthreads();

  // ctx[b,:] -> xin[b, Ee + d] (bf16), and embed gather -> xin[b, 0:Ee]
  bf16_t* xrow = xin + (size_t)b * (Ee + Dimg);
  for (int d = tid; d < Dimg; d += 256) {
    const float* ip = img_seq + (size_t)b * Nimg * Dimg + d;
    float s = 0.f;
    for (int n = 0; n < Nimg; ++n) s += sc[n] * ip[(size_t)n * Dimg];
    xrow[Ee + d] = (bf16_t)s;
  }
  const int tok = tokens[b * Tsteps + t];
  const float* er = embed + (size_t)tok * Ee;
  for (int e = tid; e < Ee; e += 256) xrow[e] = (bf16_t)er[e];
}

// ---------------- LSTM pointwise update ----------------
__global__ void lstm_update_kernel(const float* __restrict__ gates,
                                   float* __restrict__ h, float* __restrict__ c,
                                   bf16_t* __restrict__ h_bf) {
  int idx = blockIdx.x * blockDim.x + threadIdx.x;   // over B*Hh
  int b = idx >> 10, u = idx & (Hh - 1);
  const float* gb = gates + (size_t)b * 4 * Hh;
  const float gi = gb[u];
  const float gf = gb[Hh + u];
  const float gg = gb[2 * Hh + u];
  const float go = gb[3 * Hh + u];
  const float si = 1.f / (1.f + __expf(-gi));
  const float sf = 1.f / (1.f + __expf(-gf));
  const float so = 1.f / (1.f + __expf(-go));
  const float cn = sf * c[idx] + si * tanhf(gg);
  const float hn = so * tanhf(cn);
  c[idx] = cn;
  h[idx] = hn;
  h_bf[idx] = (bf16_t)hn;
}

// ---------------- host orchestration ----------------
extern "C" void kernel_launch(void* const* d_in, const int* in_sizes, int n_in,
                              void* d_out, int out_size, void* d_ws, size_t ws_size,
                              hipStream_t stream) {
  (void)in_sizes; (void)n_in; (void)out_size; (void)ws_size;
  const float* img_seq = (const float*)d_in[0];
  const int*   tokens  = (const int*)d_in[1];
  const float* embed   = (const float*)d_in[2];
  const float* W_head1 = (const float*)d_in[3];
  const float* W_ih    = (const float*)d_in[4];
  const float* W_hh    = (const float*)d_in[5];
  const float* b_ih    = (const float*)d_in[6];
  const float* b_hh    = (const float*)d_in[7];
  const float* W_head  = (const float*)d_in[8];
  const float* W_img   = (const float*)d_in[9];
  const float* W_hid   = (const float*)d_in[10];
  const float* w_score = (const float*)d_in[11];
  const float* W_h0    = (const float*)d_in[12];
  const float* b_h0    = (const float*)d_in[13];
  const float* W_c0    = (const float*)d_in[14];
  const float* b_c0    = (const float*)d_in[15];
  float* out = (float*)d_out;

  char* ws = (char*)d_ws;
  size_t off = 0;
  auto take = [&](size_t bytes) -> void* {
    void* p = ws + off;
    off = (off + bytes + 255) & ~(size_t)255;
    return p;
  };
  bf16_t* Whead_bf  = (bf16_t*)take((size_t)Vv * Hh * 2);          // 65.5 MB (L2-resident)
  bf16_t* Wih_bf    = (bf16_t*)take((size_t)4 * Hh * Ee * 2);
  bf16_t* Whh_bf    = (bf16_t*)take((size_t)4 * Hh * Hh * 2);
  bf16_t* Whead1_bf = (bf16_t*)take((size_t)Ee * (Ee + Dimg) * 2);
  bf16_t* Wimg_bf   = (bf16_t*)take((size_t)Aa * Dimg * 2);
  bf16_t* img_bf    = (bf16_t*)take((size_t)Bdim * Nimg * Dimg * 2);
  float*  proj_img  = (float*) take((size_t)Bdim * Nimg * Aa * 4);
  float*  g_mean    = (float*) take((size_t)Bdim * Dimg * 4);
  float*  h_st      = (float*) take((size_t)Bdim * Hh * 4);
  float*  c_st      = (float*) take((size_t)Bdim * Hh * 4);
  bf16_t* h_bf      = (bf16_t*)take((size_t)Bdim * Hh * 2);
  bf16_t* xin       = (bf16_t*)take((size_t)Bdim * (Ee + Dimg) * 2);
  float*  x_f       = (float*) take((size_t)Bdim * Ee * 4);
  bf16_t* x_bf      = (bf16_t*)take((size_t)Bdim * Ee * 2);
  float*  gates     = (float*) take((size_t)Bdim * 4 * Hh * 4);
  float*  bias_sum  = (float*) take((size_t)4 * Hh * 4);

  auto cvt = [&](const float* s, bf16_t* d, size_t n) {
    f32_to_bf16_kernel<<<dim3((unsigned)((n + 255) / 256)), 256, 0, stream>>>(s, d, (int)n);
  };
  auto gemm = [&](const bf16_t* A, int lda, const bf16_t* Bm, int ldb,
                  float* oF, int ldo, bf16_t* oBF, int ldobf,
                  const float* bias, int M, int N, int K, int accum) {
    dim3 grid((unsigned)(N / 64), (unsigned)((M / 16 + 3) / 4));
    wmma_gemm_bf16_kernel<<<grid, 128, 0, stream>>>(A, lda, Bm, ldb, oF, ldo,
                                                    oBF, ldobf, bias, M, N, K, accum);
  };

  // one-time: bf16 weight/data conversion
  cvt(W_head,  Whead_bf,  (size_t)Vv * Hh);
  cvt(W_ih,    Wih_bf,    (size_t)4 * Hh * Ee);
  cvt(W_hh,    Whh_bf,    (size_t)4 * Hh * Hh);
  cvt(W_head1, Whead1_bf, (size_t)Ee * (Ee + Dimg));
  cvt(W_img,   Wimg_bf,   (size_t)Aa * Dimg);
  cvt(img_seq, img_bf,    (size_t)Bdim * Nimg * Dimg);

  // one-time: g = mean, h0/c0, bias_sum
  mean_kernel<<<(Bdim * Dimg) / 256, 256, 0, stream>>>(img_seq, g_mean);
  init_hc_kernel<<<dim3((Bdim * Hh) / 256, 2), 256, 0, stream>>>(
      g_mean, W_h0, b_h0, W_c0, b_c0, h_st, c_st, h_bf);
  bias_sum_kernel<<<(4 * Hh) / 256, 256, 0, stream>>>(b_ih, b_hh, bias_sum, 4 * Hh);

  // one-time: proj_img[BN, A] = img_bf[BN, D] @ Wimg^T  (WMMA, 13 GFLOP)
  gemm(img_bf, Dimg, Wimg_bf, Dimg, proj_img, Aa, nullptr, 0, nullptr,
       Bdim * Nimg, Aa, Dimg, 0);

  // sequential decode
  for (int t = 0; t < Tsteps; ++t) {
    // attention + ctx + embed gather -> xin (bf16 [B, E+D])
    attn_ctx_kernel<<<Bdim, 256, 0, stream>>>(h_st, W_hid, w_score, proj_img,
                                              img_seq, embed, tokens, t, xin);
    // x[B,E] = xin @ W_head1^T
    gemm(xin, Ee + Dimg, Whead1_bf, Ee + Dimg, x_f, Ee, x_bf, Ee, nullptr,
         Bdim, Ee, Ee + Dimg, 0);
    // gates = x @ W_ih^T + (b_ih + b_hh)
    gemm(x_bf, Ee, Wih_bf, Ee, gates, 4 * Hh, nullptr, 0, bias_sum,
         Bdim, 4 * Hh, Ee, 0);
    // gates += h @ W_hh^T
    gemm(h_bf, Hh, Whh_bf, Hh, gates, 4 * Hh, nullptr, 0, nullptr,
         Bdim, 4 * Hh, Hh, 1);
    // LSTM pointwise -> h, c, h_bf
    lstm_update_kernel<<<(Bdim * Hh) / 256, 256, 0, stream>>>(gates, h_st, c_st, h_bf);
    // logits[B, V] = h_new @ W_head^T  -> out[:, t, :]  (dominant GEMM)
    gemm(h_bf, Hh, Whead_bf, Hh, out + (size_t)t * Vv, Tsteps * Vv, nullptr, 0,
         nullptr, Bdim, Vv, Hh, 0);
  }
}
